// XbmInBatchTripletLoss_39779987096333
// MI455X (gfx1250) — matrix operand order repaired and checked by reading, not compiled, not run on hardware
//
#include <hip/hip_runtime.h>
#include <math.h>

// ---------------------------------------------------------------------------
// XBM in-batch triplet loss, fused GEMM + epilogue for MI455X (gfx1250).
// 8192x8192x512 GEMM (68.7 GFLOP) dominates; inputs (48MB) are L2-resident.
// v2: one-time fp32 -> bf16(hi/lo) conversion pre-pass (removes the 128x
// redundant per-tile conversion that made v1 VALU-bound), 128x128 block tile
// (8 waves x 24 v_wmma_f32_16x16x32_bf16 per K-step), global_prefetch of the
// next K slab. Split-bf16 (hi*hi + hi*lo + lo*hi) keeps ~fp32 accuracy.
// ---------------------------------------------------------------------------

#define NROWS 8192
#define DIM   512
#define EPSF  1e-6f

#define LDSTR 40   // ushorts per LDS tile row: 32 data + 8 pad (80B, 16B aligned)

typedef __attribute__((ext_vector_type(16))) __bf16 v16bf;
typedef __attribute__((ext_vector_type(8)))  __bf16 v8bf;
typedef __attribute__((ext_vector_type(8)))  float  v8f;

__device__ __forceinline__ unsigned short f2bf_rne(float x) {
    unsigned u = __float_as_uint(x);
    unsigned r = u + 0x7fffu + ((u >> 16) & 1u);
    return (unsigned short)(r >> 16);
}
__device__ __forceinline__ float bf2f(unsigned short h) {
    return __uint_as_float(((unsigned)h) << 16);
}

// Two contiguous 16-byte LDS reads -> one v16bf fragment.
__device__ __forceinline__ v16bf ld_frag(const unsigned short* p0,
                                         const unsigned short* p1) {
    v8bf c0 = *reinterpret_cast<const v8bf*>(p0);
    v8bf c1 = *reinterpret_cast<const v8bf*>(p1);
    return __builtin_shufflevector(c0, c1, 0, 1, 2, 3, 4, 5, 6, 7,
                                           8, 9, 10, 11, 12, 13, 14, 15);
}

// ---------------------------------------------------------------------------
// Per-row statistics: sa, a2, sp, p2, d_pos, d_neg_orig
// ---------------------------------------------------------------------------
__global__ void __launch_bounds__(256)
stats_kernel(const float* __restrict__ anchor, const float* __restrict__ pos,
             const float* __restrict__ neg,
             float* __restrict__ sa, float* __restrict__ a2,
             float* __restrict__ sp, float* __restrict__ p2,
             float* __restrict__ dpos, float* __restrict__ dnegd) {
    const int row = blockIdx.x;
    const int tid = threadIdx.x;
    const float* ar = anchor + (size_t)row * DIM;
    const float* pr = pos    + (size_t)row * DIM;
    const float* nr = neg    + (size_t)row * DIM;
    float s_a = 0.f, s_a2 = 0.f, s_p = 0.f, s_p2 = 0.f, s_dp = 0.f, s_dn = 0.f;
    for (int k = tid; k < DIM; k += 256) {
        float a = ar[k], p = pr[k], n = nr[k];
        s_a += a;  s_a2 += a * a;
        s_p += p;  s_p2 += p * p;
        float dp = a - p + EPSF; s_dp += dp * dp;
        float dn = a - n + EPSF; s_dn += dn * dn;
    }
    __shared__ float sm[6][256];
    sm[0][tid] = s_a;  sm[1][tid] = s_a2;
    sm[2][tid] = s_p;  sm[3][tid] = s_p2;
    sm[4][tid] = s_dp; sm[5][tid] = s_dn;
    __syncthreads();
    for (int s = 128; s > 0; s >>= 1) {
        if (tid < s) {
            #pragma unroll
            for (int q = 0; q < 6; ++q) sm[q][tid] += sm[q][tid + s];
        }
        __syncthreads();
    }
    if (tid == 0) {
        sa[row]    = sm[0][0];
        a2[row]    = sm[1][0];
        sp[row]    = sm[2][0];
        p2[row]    = sm[3][0];
        dpos[row]  = sqrtf(sm[4][0]);
        dnegd[row] = sqrtf(sm[5][0]);
    }
}

// ---------------------------------------------------------------------------
// One-time fp32 -> bf16 hi/lo conversion of anchor and pos (8 floats/thread).
// ---------------------------------------------------------------------------
__device__ __forceinline__ void cvt_pack8(float4 v0, float4 v1,
                                          uint4& hp, uint4& lp) {
    float f[8] = {v0.x, v0.y, v0.z, v0.w, v1.x, v1.y, v1.z, v1.w};
    unsigned short h[8], l[8];
    #pragma unroll
    for (int e = 0; e < 8; ++e) {
        h[e] = f2bf_rne(f[e]);
        l[e] = f2bf_rne(f[e] - bf2f(h[e]));
    }
    hp.x = (unsigned)h[0] | ((unsigned)h[1] << 16);
    hp.y = (unsigned)h[2] | ((unsigned)h[3] << 16);
    hp.z = (unsigned)h[4] | ((unsigned)h[5] << 16);
    hp.w = (unsigned)h[6] | ((unsigned)h[7] << 16);
    lp.x = (unsigned)l[0] | ((unsigned)l[1] << 16);
    lp.y = (unsigned)l[2] | ((unsigned)l[3] << 16);
    lp.z = (unsigned)l[4] | ((unsigned)l[5] << 16);
    lp.w = (unsigned)l[6] | ((unsigned)l[7] << 16);
}

__global__ void __launch_bounds__(256)
convert_kernel(const float* __restrict__ anchor, const float* __restrict__ pos,
               unsigned short* __restrict__ Ahi_g, unsigned short* __restrict__ Alo_g,
               unsigned short* __restrict__ Bhi_g, unsigned short* __restrict__ Blo_g) {
    size_t g = (size_t)blockIdx.x * 256 + threadIdx.x;   // one per 8 elements
    size_t base = g * 8;
    float4 a0 = *reinterpret_cast<const float4*>(anchor + base);
    float4 a1 = *reinterpret_cast<const float4*>(anchor + base + 4);
    float4 p0 = *reinterpret_cast<const float4*>(pos + base);
    float4 p1 = *reinterpret_cast<const float4*>(pos + base + 4);
    uint4 hp, lp;
    cvt_pack8(a0, a1, hp, lp);
    *reinterpret_cast<uint4*>(Ahi_g + base) = hp;
    *reinterpret_cast<uint4*>(Alo_g + base) = lp;
    cvt_pack8(p0, p1, hp, lp);
    *reinterpret_cast<uint4*>(Bhi_g + base) = hp;
    *reinterpret_cast<uint4*>(Blo_g + base) = lp;
}

// ---------------------------------------------------------------------------
// Shared epilogue: distance -> log1p(exp) for one accumulator element.
// ---------------------------------------------------------------------------
__device__ __forceinline__ float loss_elem(float dot, int i, int j,
                                           float ai2, float sai, float dpi, float dni,
                                           float p2j, float spj) {
    float sq = ai2 + p2j - 2.0f * dot
             + 2.0f * EPSF * (sai - spj)
             + (float)DIM * EPSF * EPSF;
    float dist = sqrtf(fmaxf(sq, 1e-12f));
    float dn   = (i == j) ? dni : dist;
    return log1pf(__expf(dpi - dn));
}

// ---------------------------------------------------------------------------
// Fast path: 128x128 tile, 256 threads (8 waves), bf16 hi/lo pre-converted.
// Each wave owns 32x64 = 2x4 tiles of v_wmma_f32_16x16x32_bf16.
// ---------------------------------------------------------------------------
__global__ void __launch_bounds__(256)
triplet_gemm128_kernel(const unsigned short* __restrict__ Ahi_g,
                       const unsigned short* __restrict__ Alo_g,
                       const unsigned short* __restrict__ Bhi_g,
                       const unsigned short* __restrict__ Blo_g,
                       const float* __restrict__ sa, const float* __restrict__ a2,
                       const float* __restrict__ sp, const float* __restrict__ p2,
                       const float* __restrict__ dpos, const float* __restrict__ dnegd,
                       double* __restrict__ gacc) {
    __shared__ unsigned short Ahi[128 * LDSTR], Alo[128 * LDSTR];
    __shared__ unsigned short Bhi[128 * LDSTR], Blo[128 * LDSTR];
    __shared__ float red[256];

    const int tid  = threadIdx.x;
    const int wave = tid >> 5;
    const int lane = tid & 31;
    const int h    = lane >> 4;
    const int ln   = lane & 15;
    const int wm   = (wave >> 1) * 32;   // 4 wave-rows of 32
    const int wn   = (wave & 1) * 64;    // 2 wave-cols of 64
    const int ibase = blockIdx.y * 128;
    const int jbase = blockIdx.x * 128;

    // Staging assignment: 512 16B-chunks per (matrix,half); thread does 2.
    // slot s (0..511): row = s>>2 (0..127), c = s&3 (16B chunk within 64B row)
    const int s0   = tid;         // slots tid and tid+256
    const int row0 = s0 >> 2, c0 = s0 & 3;
    const int s1   = tid + 256;
    const int row1 = s1 >> 2, c1 = s1 & 3;

    v8f acc[2][4] = {};

    for (int kb = 0; kb < DIM; kb += 32) {
        // ---- prefetch next K slab into caches (global_prefetch_b8) ----
        if (kb + 32 < DIM) {
            size_t na0 = (size_t)(ibase + row0) * DIM + (kb + 32) + c0 * 8;
            size_t nb0 = (size_t)(jbase + row0) * DIM + (kb + 32) + c0 * 8;
            __builtin_prefetch(Ahi_g + na0, 0, 0);
            __builtin_prefetch(Alo_g + na0, 0, 0);
            __builtin_prefetch(Bhi_g + nb0, 0, 0);
            __builtin_prefetch(Blo_g + nb0, 0, 0);
        }

        // ---- stage bf16 tiles global -> LDS (pure byte movement) ----
        {
            size_t ga0 = (size_t)(ibase + row0) * DIM + kb + c0 * 8;
            size_t ga1 = (size_t)(ibase + row1) * DIM + kb + c1 * 8;
            size_t gb0 = (size_t)(jbase + row0) * DIM + kb + c0 * 8;
            size_t gb1 = (size_t)(jbase + row1) * DIM + kb + c1 * 8;
            int la0 = row0 * LDSTR + c0 * 8;
            int la1 = row1 * LDSTR + c1 * 8;
            *reinterpret_cast<uint4*>(Ahi + la0) = *reinterpret_cast<const uint4*>(Ahi_g + ga0);
            *reinterpret_cast<uint4*>(Ahi + la1) = *reinterpret_cast<const uint4*>(Ahi_g + ga1);
            *reinterpret_cast<uint4*>(Alo + la0) = *reinterpret_cast<const uint4*>(Alo_g + ga0);
            *reinterpret_cast<uint4*>(Alo + la1) = *reinterpret_cast<const uint4*>(Alo_g + ga1);
            *reinterpret_cast<uint4*>(Bhi + la0) = *reinterpret_cast<const uint4*>(Bhi_g + gb0);
            *reinterpret_cast<uint4*>(Bhi + la1) = *reinterpret_cast<const uint4*>(Bhi_g + gb1);
            *reinterpret_cast<uint4*>(Blo + la0) = *reinterpret_cast<const uint4*>(Blo_g + gb0);
            *reinterpret_cast<uint4*>(Blo + la1) = *reinterpret_cast<const uint4*>(Blo_g + gb1);
        }
        __syncthreads();

        // ---- fragments (ISA 16-bit A/B wave32 layouts) ----
        v16bf ahf[2], alf[2], bhf[4], blf[4];
        #pragma unroll
        for (int t = 0; t < 2; ++t) {
            int m  = wm + t * 16 + ln;
            int oa = m * LDSTR + h * 8;
            ahf[t] = ld_frag(Ahi + oa, Ahi + oa + 16);
            alf[t] = ld_frag(Alo + oa, Alo + oa + 16);
        }
        #pragma unroll
        for (int u = 0; u < 4; ++u) {
            int n  = wn + u * 16 + ln;
            int ob = n * LDSTR + h * 8;
            bhf[u] = ld_frag(Bhi + ob, Bhi + ob + 16);
            blf[u] = ld_frag(Blo + ob, Blo + ob + 16);
        }

        // ---- split-precision product: hi*hi + hi*lo + lo*hi ----
        #pragma unroll
        for (int t = 0; t < 2; ++t) {
            #pragma unroll
            for (int u = 0; u < 4; ++u) {
                acc[t][u] = __builtin_amdgcn_wmma_f32_16x16x32_bf16(
                    false, ahf[t], false, bhf[u], (short)0, acc[t][u], false, false);
                acc[t][u] = __builtin_amdgcn_wmma_f32_16x16x32_bf16(
                    false, ahf[t], false, blf[u], (short)0, acc[t][u], false, false);
                acc[t][u] = __builtin_amdgcn_wmma_f32_16x16x32_bf16(
                    false, alf[t], false, bhf[u], (short)0, acc[t][u], false, false);
            }
        }
        __syncthreads();
    }

    // ---- fused epilogue ----
    float p2j[4], spj[4];
    int jv[4];
    #pragma unroll
    for (int u = 0; u < 4; ++u) {
        jv[u]  = jbase + wn + u * 16 + ln;
        p2j[u] = p2[jv[u]];
        spj[u] = sp[jv[u]];
    }
    float partial = 0.f;
    #pragma unroll
    for (int t = 0; t < 2; ++t) {
        #pragma unroll
        for (int r = 0; r < 8; ++r) {
            int i = ibase + wm + t * 16 + r + 8 * h;
            float ai2 = a2[i], sai = sa[i], dpi = dpos[i], dni = dnegd[i];
            #pragma unroll
            for (int u = 0; u < 4; ++u)
                partial += loss_elem(acc[t][u][r], i, jv[u],
                                     ai2, sai, dpi, dni, p2j[u], spj[u]);
        }
    }
    red[tid] = partial;
    __syncthreads();
    for (int s = 128; s > 0; s >>= 1) {
        if (tid < s) red[tid] += red[tid + s];
        __syncthreads();
    }
    if (tid == 0) atomicAdd(gacc, (double)red[0]);
}

// ---------------------------------------------------------------------------
// Fallback path (small ws): 64x64 tile, in-kernel conversion (v1 kernel).
// ---------------------------------------------------------------------------
__device__ __forceinline__ void cvt_store4(unsigned short* hi, unsigned short* lo,
                                           int off, float4 v) {
    unsigned short h0 = f2bf_rne(v.x), h1 = f2bf_rne(v.y);
    unsigned short h2 = f2bf_rne(v.z), h3 = f2bf_rne(v.w);
    unsigned short l0 = f2bf_rne(v.x - bf2f(h0)), l1 = f2bf_rne(v.y - bf2f(h1));
    unsigned short l2 = f2bf_rne(v.z - bf2f(h2)), l3 = f2bf_rne(v.w - bf2f(h3));
    uint2 hp, lp;
    hp.x = (unsigned)h0 | ((unsigned)h1 << 16);
    hp.y = (unsigned)h2 | ((unsigned)h3 << 16);
    lp.x = (unsigned)l0 | ((unsigned)l1 << 16);
    lp.y = (unsigned)l2 | ((unsigned)l3 << 16);
    *reinterpret_cast<uint2*>(hi + off) = hp;
    *reinterpret_cast<uint2*>(lo + off) = lp;
}

__global__ void __launch_bounds__(128)
triplet_gemm_conv_kernel(const float* __restrict__ anchor, const float* __restrict__ pos,
                         const float* __restrict__ sa, const float* __restrict__ a2,
                         const float* __restrict__ sp, const float* __restrict__ p2,
                         const float* __restrict__ dpos, const float* __restrict__ dnegd,
                         double* __restrict__ gacc) {
    __shared__ unsigned short Ahi[64 * LDSTR], Alo[64 * LDSTR];
    __shared__ unsigned short Bhi[64 * LDSTR], Blo[64 * LDSTR];
    __shared__ float red[128];

    const int tid  = threadIdx.x;
    const int wave = tid >> 5;
    const int lane = tid & 31;
    const int h    = lane >> 4;
    const int ln   = lane & 15;
    const int wm   = (wave >> 1) * 32;
    const int wn   = (wave & 1) * 32;
    const int ibase = blockIdx.y * 64;
    const int jbase = blockIdx.x * 64;

    v8f acc[2][2] = {};

    for (int kb = 0; kb < DIM; kb += 32) {
        #pragma unroll
        for (int q = 0; q < 4; ++q) {
            int idx = tid + q * 128;
            int row = idx >> 3;
            int c4  = idx & 7;
            float4 av = *reinterpret_cast<const float4*>(
                anchor + (size_t)(ibase + row) * DIM + kb + c4 * 4);
            float4 pv = *reinterpret_cast<const float4*>(
                pos + (size_t)(jbase + row) * DIM + kb + c4 * 4);
            int off = row * LDSTR + c4 * 4;
            cvt_store4(Ahi, Alo, off, av);
            cvt_store4(Bhi, Blo, off, pv);
        }
        __syncthreads();

        v16bf ahf[2], alf[2], bhf[2], blf[2];
        #pragma unroll
        for (int t = 0; t < 2; ++t) {
            int m  = wm + t * 16 + ln;
            int oa = m * LDSTR + h * 8;
            ahf[t] = ld_frag(Ahi + oa, Ahi + oa + 16);
            alf[t] = ld_frag(Alo + oa, Alo + oa + 16);
            int n  = wn + t * 16 + ln;
            int ob = n * LDSTR + h * 8;
            bhf[t] = ld_frag(Bhi + ob, Bhi + ob + 16);
            blf[t] = ld_frag(Blo + ob, Blo + ob + 16);
        }

        #pragma unroll
        for (int t = 0; t < 2; ++t) {
            #pragma unroll
            for (int u = 0; u < 2; ++u) {
                acc[t][u] = __builtin_amdgcn_wmma_f32_16x16x32_bf16(
                    false, ahf[t], false, bhf[u], (short)0, acc[t][u], false, false);
                acc[t][u] = __builtin_amdgcn_wmma_f32_16x16x32_bf16(
                    false, ahf[t], false, blf[u], (short)0, acc[t][u], false, false);
                acc[t][u] = __builtin_amdgcn_wmma_f32_16x16x32_bf16(
                    false, alf[t], false, bhf[u], (short)0, acc[t][u], false, false);
            }
        }
        __syncthreads();
    }

    float partial = 0.f;
    #pragma unroll
    for (int t = 0; t < 2; ++t) {
        #pragma unroll
        for (int r = 0; r < 8; ++r) {
            int i = ibase + wm + t * 16 + r + 8 * h;
            float ai2 = a2[i], sai = sa[i], dpi = dpos[i], dni = dnegd[i];
            #pragma unroll
            for (int u = 0; u < 2; ++u) {
                int j = jbase + wn + u * 16 + ln;
                partial += loss_elem(acc[t][u][r], i, j,
                                     ai2, sai, dpi, dni, p2[j], sp[j]);
            }
        }
    }
    red[tid] = partial;
    __syncthreads();
    for (int s = 64; s > 0; s >>= 1) {
        if (tid < s) red[tid] += red[tid + s];
        __syncthreads();
    }
    if (tid == 0) atomicAdd(gacc, (double)red[0]);
}

__global__ void finalize_kernel(const double* __restrict__ gacc,
                                float* __restrict__ out) {
    out[0] = (float)(gacc[0] / ((double)NROWS * (double)NROWS));
}

// ---------------------------------------------------------------------------
extern "C" void kernel_launch(void* const* d_in, const int* in_sizes, int n_in,
                              void* d_out, int out_size, void* d_ws, size_t ws_size,
                              hipStream_t stream) {
    const float* anchor = (const float*)d_in[0];
    const float* pos    = (const float*)d_in[1];
    const float* neg    = (const float*)d_in[2];

    char* ws = (char*)d_ws;
    double* gacc = (double*)ws;                  // 8B accumulator (offset 0)
    float* sa    = (float*)(ws + 256);
    float* a2    = sa    + NROWS;
    float* sp    = a2    + NROWS;
    float* p2    = sp    + NROWS;
    float* dpos  = p2    + NROWS;
    float* dnegd = dpos  + NROWS;

    const size_t statsBytes = 256 + 6 * sizeof(float) * NROWS;   // 196,864 (256-aligned)
    const size_t convBytes  = (size_t)NROWS * DIM * sizeof(unsigned short); // 8MB / array

    hipMemsetAsync(d_ws, 0, 256, stream);        // zero the accumulator (graph-safe)

    stats_kernel<<<NROWS, 256, 0, stream>>>(anchor, pos, neg,
                                            sa, a2, sp, p2, dpos, dnegd);

    if (ws_size >= statsBytes + 4 * convBytes) {
        // Fast path: pre-convert to bf16 hi/lo, then 128x128 WMMA tiles.
        unsigned short* Ahi_g = (unsigned short*)(ws + statsBytes);
        unsigned short* Alo_g = (unsigned short*)(ws + statsBytes + 1 * convBytes);
        unsigned short* Bhi_g = (unsigned short*)(ws + statsBytes + 2 * convBytes);
        unsigned short* Blo_g = (unsigned short*)(ws + statsBytes + 3 * convBytes);

        const int nconv = (NROWS * DIM / 8) / 256;   // 2048 blocks
        convert_kernel<<<nconv, 256, 0, stream>>>(anchor, pos,
                                                  Ahi_g, Alo_g, Bhi_g, Blo_g);

        dim3 grid(NROWS / 128, NROWS / 128);
        triplet_gemm128_kernel<<<grid, 256, 0, stream>>>(Ahi_g, Alo_g, Bhi_g, Blo_g,
                                                         sa, a2, sp, p2, dpos, dnegd,
                                                         gacc);
    } else {
        // Fallback: in-kernel conversion, 64x64 tiles.
        dim3 grid(NROWS / 64, NROWS / 64);
        triplet_gemm_conv_kernel<<<grid, 128, 0, stream>>>(anchor, pos,
                                                           sa, a2, sp, p2, dpos, dnegd,
                                                           gacc);
    }

    finalize_kernel<<<1, 1, 0, stream>>>(gacc, (float*)d_out);
}
